// Timeception_75376676045197
// MI455X (gfx1250) — matrix operand order repaired and checked by reading, not compiled, not run on hardware
//
#include <hip/hip_runtime.h>

// ---------------------------------------------------------------------------
// Timeception (2 stacked TC layers) for MI455X / gfx1250, bf16 WMMA path.
// Grouped 1x1 conv == per-group GEMM -> v_wmma_f32_16x16x32_bf16.
// One workgroup per (b, s, g); full T range owned -> temporal branch ops,
// BN, ReLU, shuffle and stride-2 temporal maxpool fused through LDS.
// Round 3: 4 waves with 2-D (M,N) wave tiling sized to avoid VGPR spills
// (acc <= 80 VGPRs/wave) + __launch_bounds__ so the compiler budgets
// registers for the real 128-thread block. ds_load_b128 fragments, padded
// LDS strides kept from round 2.
// ---------------------------------------------------------------------------

typedef __attribute__((ext_vector_type(16))) __bf16 v16bf;
typedef __attribute__((ext_vector_type(8)))  float  v8f;

union Frag {
    v16bf v;
    uint4 q[2];      // two 16B chunks -> ds_load_b128 pairs
};

__device__ __forceinline__ unsigned short f2bf(float f) {
    unsigned int u = __float_as_uint(f);
    unsigned int r = u + 0x7fffu + ((u >> 16) & 1u);   // round-to-nearest-even
    return (unsigned short)(r >> 16);
}
__device__ __forceinline__ float bf2f(unsigned short h) {
    return __uint_as_float(((unsigned int)h) << 16);
}

// ------------------------- problem constants -------------------------------
#define BB   16
#define GG   8

// layer 0
#define T0    64
#define S0    49
#define C0    1024
#define CG0   128
#define LDA0  136        // padded LDS row stride (bf16 elems): 272B rows
#define NB0   32
#define NP0   160        // padded N = NWN * NTW * 16
#define NTW0  5          // N tiles per wave
#define NWN0  2          // waves in N
#define MTW0  2          // M tiles per wave
#define NWM0  2          // waves in M   (MTW*NWM = T/16)
#define KS0   4          // K steps of 32
// layer 0 output / layer 1 input
#define T1    32
#define S1    51
#define C1    1280
#define CG1   160
#define LDA1  168        // 336B rows
#define NB1   40
#define N51   200
#define NP1   224
#define NTW1  7
#define NWN1  2
#define MTW1  1
#define NWM1  2
#define KS1   5

// ------------------------- workspace layout (bytes) ------------------------
static constexpr size_t OFF_W0    = 0;
static constexpr size_t SZ_W0     = (size_t)GG * NP0 * CG0 * 2;       // 327680
static constexpr size_t OFF_W1    = OFF_W0 + SZ_W0;
static constexpr size_t SZ_W1     = (size_t)GG * NP1 * CG1 * 2;       // 573440
static constexpr size_t OFF_P0    = OFF_W1 + SZ_W1;
static constexpr size_t SZ_P0     = (size_t)3 * GG * NP0 * 4;
static constexpr size_t OFF_P1    = OFF_P0 + SZ_P0;
static constexpr size_t OFF_POOL0 = 1048576;
static constexpr size_t SZ_POOL0  = (size_t)BB * T0 * 2 * C0 * 2;     // 4 MB
static constexpr size_t OFF_MID   = OFF_POOL0 + SZ_POOL0;
static constexpr size_t SZ_MID    = (size_t)BB * T1 * S1 * C1 * 2;    // ~64 MB
static constexpr size_t OFF_POOL1 = OFF_MID + SZ_MID;

// ---------------------------------------------------------------------------
// Prep: permute weights (5,G,Cg,nb) f32 -> (G, Npad, Cg) bf16 (row = p*nb+n),
// and fold bias / BN into per-(g, nIdx) bias, scale, shift arrays.
// ---------------------------------------------------------------------------
__global__ void prep_params_kernel(const float* __restrict__ w,
                                   const float* __restrict__ b,
                                   const float* __restrict__ gam,
                                   const float* __restrict__ bet,
                                   const float* __restrict__ mu,
                                   const float* __restrict__ var,
                                   unsigned short* __restrict__ wbf,
                                   float* __restrict__ prm,
                                   int Cg, int NB_, int NPAD) {
    int row = blockIdx.x * blockDim.x + threadIdx.x;
    int rows = GG * NPAD;
    if (row >= rows) return;
    int g = row / NPAD, nIdx = row % NPAD;
    unsigned short* dstW = wbf + (size_t)row * Cg;
    float* biasA = prm;
    float* scA   = prm + rows;
    float* shA   = prm + 2 * rows;
    if (nIdx < 5 * NB_) {
        int p = nIdx / NB_, n = nIdx % NB_;
        for (int k = 0; k < Cg; ++k)
            dstW[k] = f2bf(w[((size_t)(p * GG + g) * Cg + k) * NB_ + n]);
        int pi = (p * GG + g) * NB_ + n;
        float sc = gam[pi] * rsqrtf(var[pi] + 1e-3f);
        biasA[row] = b[pi];
        scA[row]   = sc;
        shA[row]   = bet[pi] - mu[pi] * sc;
    } else {
        for (int k = 0; k < Cg; ++k) dstW[k] = 0;
        biasA[row] = 0.f; scA[row] = 0.f; shA[row] = 0.f;
    }
}

// ---------------------------------------------------------------------------
// Spatial max/mean pooling -> the two augmented rows (bf16 output).
// ---------------------------------------------------------------------------
__global__ void pool_f32_kernel(const float* __restrict__ x,
                                unsigned short* __restrict__ pool,
                                int BT, int S, int C) {
    size_t i = (size_t)blockIdx.x * blockDim.x + threadIdx.x;
    if (i >= (size_t)BT * C) return;
    size_t bt = i / C; int c = (int)(i % C);
    const float* base = x + bt * (size_t)S * C + c;
    float mx = -3.4e38f, sm = 0.f;
    for (int s = 0; s < S; ++s) {
        float v = base[(size_t)s * C];
        mx = fmaxf(mx, v); sm += v;
    }
    pool[(bt * 2 + 0) * C + c] = f2bf(mx);
    pool[(bt * 2 + 1) * C + c] = f2bf(sm / (float)S);
}

__global__ void pool_bf16_kernel(const unsigned short* __restrict__ x,
                                 unsigned short* __restrict__ pool,
                                 int BT, int S, int C) {
    size_t i = (size_t)blockIdx.x * blockDim.x + threadIdx.x;
    if (i >= (size_t)BT * C) return;
    size_t bt = i / C; int c = (int)(i % C);
    const unsigned short* base = x + bt * (size_t)S * C + c;
    float mx = -3.4e38f, sm = 0.f;
    for (int s = 0; s < S; ++s) {
        float v = bf2f(base[(size_t)s * C]);
        mx = fmaxf(mx, v); sm += v;
    }
    pool[(bt * 2 + 0) * C + c] = f2bf(mx);
    pool[(bt * 2 + 1) * C + c] = f2bf(sm / (float)S);
}

// ---------------------------------------------------------------------------
// Fused TC layer. NWM*NWN waves per block; wave (wm, wn) owns MTW M-tiles and
// NTW N-tiles, so each B fragment feeds MTW WMMAs and acc stays <= 80 VGPRs.
//   T    : input time steps (MTW*NWM*16)
//   SIN  : input spatial rows (augmented rows come from `pool`)
//   CG   : K (channels per group), = KS*32;  LDA = CG+8 (padded LDS stride)
//   NPAD : NWN*NTW*16 >= 5*NBc
// ---------------------------------------------------------------------------
template <int T, int SIN, int CG, int LDA, int NBc, int NPAD, int CIN,
          int NTW, int NWN, int MTW, int NWM, int KS, bool SRC_BF, bool OUT_BF>
__global__ void
__launch_bounds__(32 * NWM * NWN, 1)
tc_layer_kernel(const float* __restrict__ xf,
                const unsigned short* __restrict__ xb,
                const unsigned short* __restrict__ pool,
                const unsigned short* __restrict__ wbf,
                const float* __restrict__ prm,
                const float* __restrict__ k3,
                const float* __restrict__ k5,
                const float* __restrict__ k7,
                void* __restrict__ outp) {
    constexpr int SA = SIN + 2;
    constexpr int N5 = 5 * NBc;
    constexpr int TOUT = T / 2;

    int blk = blockIdx.x;
    int b = blk / (SA * GG);
    int rem = blk % (SA * GG);
    int s = rem / GG;
    int g = rem % GG;

    extern __shared__ char smem[];
    unsigned short* sA = (unsigned short*)smem;               // T x LDA (bf16)
    unsigned short* sW = sA + (size_t)T * LDA;                // NPAD x LDA (bf16, N-major)
    float* sY = (float*)(sW + (size_t)NPAD * LDA);            // T x NPAD (f32)

    const int tid = threadIdx.x;
    const int nthr = 32 * NWM * NWN;

    // ---- stage A tile (convert to bf16), augmented rows from pool ----
    {
        const int AQ = T * CG / 4;
        for (int i = tid; i < AQ; i += nthr) {
            int t = i / (CG / 4), k = (i % (CG / 4)) * 4;
            union { unsigned short h[4]; uint2 u; } tmp;
            if (s < SIN) {
                if constexpr (SRC_BF) {
                    const unsigned short* src =
                        xb + ((size_t)(b * T + t) * SIN + s) * CIN + g * CG + k;
                    tmp.u = *(const uint2*)src;
                } else {
                    const float* src =
                        xf + ((size_t)(b * T + t) * SIN + s) * CIN + g * CG + k;
                    float4 f = *(const float4*)src;
                    tmp.h[0] = f2bf(f.x); tmp.h[1] = f2bf(f.y);
                    tmp.h[2] = f2bf(f.z); tmp.h[3] = f2bf(f.w);
                }
            } else {
                const unsigned short* src =
                    pool + ((size_t)(b * T + t) * 2 + (s - SIN)) * CIN + g * CG + k;
                tmp.u = *(const uint2*)src;
            }
            *(uint2*)(sA + (size_t)t * LDA + k) = tmp.u;
        }
    }
    // ---- stage W tile ((g, Npad, CG) bf16 -> padded-stride LDS image) ----
    {
        const int CQ = CG / 8;                 // uint4 chunks per row
        const unsigned short* srcW = wbf + (size_t)g * NPAD * CG;
        for (int i = tid; i < NPAD * CQ; i += nthr) {
            int row = i / CQ, c = (i % CQ) * 8;
            *(uint4*)(sW + (size_t)row * LDA + c) =
                *(const uint4*)(srcW + (size_t)row * CG + c);
        }
    }
    __syncthreads();

    // ---- GEMM: wave (wm, wn) owns MTW M-tiles x NTW N-tiles ----
    const int wave = tid >> 5;
    const int wm   = wave % NWM;
    const int wn   = wave / NWM;
    const int lane = tid & 31;
    const int half = lane >> 4;
    const int mr   = lane & 15;

    v8f acc[NTW][MTW];
#pragma unroll
    for (int j = 0; j < NTW; ++j)
#pragma unroll
        for (int m = 0; m < MTW; ++m)
#pragma unroll
            for (int r = 0; r < 8; ++r) acc[j][m][r] = 0.f;

#pragma unroll
    for (int kk = 0; kk < KS; ++kk) {
        Frag aF[MTW];
#pragma unroll
        for (int m = 0; m < MTW; ++m) {
            const unsigned short* ab =
                sA + (size_t)((wm * MTW + m) * 16 + mr) * LDA + kk * 32 + half * 8;
            aF[m].q[0] = *(const uint4*)ab;        // K = base .. base+7
            aF[m].q[1] = *(const uint4*)(ab + 16); // K = base+16 .. base+23
        }
#pragma unroll
        for (int jt = 0; jt < NTW; ++jt) {
            const unsigned short* bb =
                sW + (size_t)((wn * NTW + jt) * 16 + mr) * LDA + kk * 32 + half * 8;
            Frag bF;
            bF.q[0] = *(const uint4*)bb;           // K = base .. base+7
            bF.q[1] = *(const uint4*)(bb + 16);    // K = base+16 .. base+23
#pragma unroll
            for (int m = 0; m < MTW; ++m)
                acc[jt][m] = __builtin_amdgcn_wmma_f32_16x16x32_bf16(
                    false, aF[m].v, false, bF.v, (short)0, acc[jt][m],
                    false, false);
        }
    }

    // ---- spill accumulators (+bias) to LDS Y ----
    const float* biasA = prm;
#pragma unroll
    for (int jt = 0; jt < NTW; ++jt) {
        int n = (wn * NTW + jt) * 16 + mr;
        float bv = biasA[g * NPAD + n];
#pragma unroll
        for (int m = 0; m < MTW; ++m)
#pragma unroll
            for (int r = 0; r < 8; ++r)
                sY[(size_t)((wm * MTW + m) * 16 + half * 8 + r) * NPAD + n] =
                    acc[jt][m][r] + bv;
    }
    __syncthreads();

    // ---- branches + BN + ReLU + shuffle + stride-2 temporal maxpool ----
    const float* scA = prm + GG * NPAD;
    const float* shA = prm + 2 * GG * NPAD;
    for (int idx = tid; idx < TOUT * N5; idx += nthr) {
        int to = idx / N5, nIdx = idx % N5;
        int p = nIdx / NBc, n = nIdx % NBc;
        float sc = scA[g * NPAD + nIdx];
        float sh = shA[g * NPAD + nIdx];
        float z[2];
#pragma unroll
        for (int q = 0; q < 2; ++q) {
            int t = 2 * to + q;
            float v;
            if (p == 0) {
                v = sY[(size_t)t * NPAD + nIdx];
            } else if (p == 4) {
                int t2 = (t + 1 < T) ? t + 1 : T - 1;
                v = fmaxf(sY[(size_t)t * NPAD + nIdx], sY[(size_t)t2 * NPAD + nIdx]);
            } else {
                int kl = (p == 1) ? 3 : ((p == 2) ? 5 : 7);
                const float* ker =
                    ((p == 1) ? k3 : ((p == 2) ? k5 : k7)) + (size_t)(g * NBc + n) * kl;
                int pl = (kl - 1) >> 1;
                v = 0.f;
                for (int j = 0; j < kl; ++j) {
                    int ts = t + j - pl;
                    if (ts >= 0 && ts < T) v += sY[(size_t)ts * NPAD + nIdx] * ker[j];
                }
            }
            v = v * sc + sh;
            z[q] = v > 0.f ? v : 0.f;
        }
        float val = fmaxf(z[0], z[1]);
        size_t co = (size_t)nIdx * GG + g;                       // channel shuffle
        size_t oidx = (((size_t)b * TOUT + to) * SA + s) * ((size_t)N5 * GG) + co;
        if constexpr (OUT_BF)
            ((unsigned short*)outp)[oidx] = f2bf(val);
        else
            ((float*)outp)[oidx] = val;
    }
}

// ---------------------------------------------------------------------------
extern "C" void kernel_launch(void* const* d_in, const int* in_sizes, int n_in,
                              void* d_out, int out_size, void* d_ws, size_t ws_size,
                              hipStream_t stream) {
    (void)in_sizes; (void)n_in; (void)out_size; (void)ws_size;

    const float* x    = (const float*)d_in[0];
    const float* w0   = (const float*)d_in[1];
    const float* b0   = (const float*)d_in[2];
    const float* k3_0 = (const float*)d_in[3];
    const float* k5_0 = (const float*)d_in[4];
    const float* k7_0 = (const float*)d_in[5];
    const float* g0   = (const float*)d_in[6];
    const float* be0  = (const float*)d_in[7];
    const float* mu0  = (const float*)d_in[8];
    const float* v0   = (const float*)d_in[9];
    const float* w1   = (const float*)d_in[10];
    const float* b1   = (const float*)d_in[11];
    const float* k3_1 = (const float*)d_in[12];
    const float* k5_1 = (const float*)d_in[13];
    const float* k7_1 = (const float*)d_in[14];
    const float* g1   = (const float*)d_in[15];
    const float* be1  = (const float*)d_in[16];
    const float* mu1  = (const float*)d_in[17];
    const float* v1   = (const float*)d_in[18];

    char* ws = (char*)d_ws;
    unsigned short* wbf0  = (unsigned short*)(ws + OFF_W0);
    unsigned short* wbf1  = (unsigned short*)(ws + OFF_W1);
    float*          prm0  = (float*)(ws + OFF_P0);
    float*          prm1  = (float*)(ws + OFF_P1);
    unsigned short* pool0 = (unsigned short*)(ws + OFF_POOL0);
    unsigned short* mid   = (unsigned short*)(ws + OFF_MID);
    unsigned short* pool1 = (unsigned short*)(ws + OFF_POOL1);

    // --- parameter prep (weights -> bf16 (g,n,k); BN folded) ---
    {
        int rows0 = GG * NP0;
        prep_params_kernel<<<(rows0 + 255) / 256, 256, 0, stream>>>(
            w0, b0, g0, be0, mu0, v0, wbf0, prm0, CG0, NB0, NP0);
        int rows1 = GG * NP1;
        prep_params_kernel<<<(rows1 + 255) / 256, 256, 0, stream>>>(
            w1, b1, g1, be1, mu1, v1, wbf1, prm1, CG1, NB1, NP1);
    }

    // --- layer 0: spatial pool rows, then fused GEMM+branches ---
    {
        size_t nth = (size_t)BB * T0 * C0;
        pool_f32_kernel<<<(int)((nth + 255) / 256), 256, 0, stream>>>(
            x, pool0, BB * T0, S0, C0);

        int grid = BB * (S0 + 2) * GG;              // 6528
        int thr  = 32 * NWM0 * NWN0;                // 128
        size_t lds = (size_t)T0 * LDA0 * 2 + (size_t)NP0 * LDA0 * 2 +
                     (size_t)T0 * NP0 * 4;          // 101888 B
        tc_layer_kernel<T0, S0, CG0, LDA0, NB0, NP0, C0,
                        NTW0, NWN0, MTW0, NWM0, KS0, false, true>
            <<<grid, thr, lds, stream>>>(x, nullptr, pool0, wbf0, prm0,
                                         k3_0, k5_0, k7_0, (void*)mid);
    }

    // --- layer 1 ---
    {
        size_t nth = (size_t)BB * T1 * C1;
        pool_bf16_kernel<<<(int)((nth + 255) / 256), 256, 0, stream>>>(
            mid, pool1, BB * T1, S1, C1);

        int grid = BB * (S1 + 2) * GG;              // 6784
        int thr  = 32 * NWM1 * NWN1;                // 128
        size_t lds = (size_t)T1 * LDA1 * 2 + (size_t)NP1 * LDA1 * 2 +
                     (size_t)T1 * NP1 * 4;          // 114688 B
        tc_layer_kernel<T1, S1, CG1, LDA1, NB1, NP1, C1,
                        NTW1, NWN1, MTW1, NWM1, KS1, true, false>
            <<<grid, thr, lds, stream>>>(nullptr, mid, pool1, wbf1, prm1,
                                         k3_1, k5_1, k7_1, d_out);
    }
}
